// MoEFFNTop1_137438954164
// MI455X (gfx1250) — matrix operand order repaired and checked by reading, not compiled, not run on hardware
//
#include <hip/hip_runtime.h>
#include <hip/hip_bf16.h>
#include <math.h>

// Problem constants (match reference)
#define TB   2048      // B*T tokens
#define CC   1024      // model dim
#define FF   4096      // ffn dim
#define EE   8         // experts

typedef __attribute__((ext_vector_type(16))) __bf16 v16bf;
typedef __attribute__((ext_vector_type(8)))  float  v8f;

#define NWAVES 8
#define NTHREADS 256

// LDS strides (bf16 elements), padded (+8 bf16 = 4 dwords) to spread banks
#define XS_STRIDE (CC + 8)              // 1032 bf16 (2064 B, 16B-aligned rows)
#define HS_STRIDE (FF + 8)              // 4104 bf16 (8208 B, 16B-aligned rows)
#define XS_BYTES  (16 * XS_STRIDE * 2)  // 33,024 B
#define HS_BYTES  (16 * HS_STRIDE * 2)  // 131,328 B
#define SMEM_BYTES (XS_BYTES + HS_BYTES) // 164,352 B  (<= 320KB WGP LDS)

__device__ __forceinline__ v8f wmma_bf16(v16bf a, v16bf b, v8f c) {
    return __builtin_amdgcn_wmma_f32_16x16x32_bf16(false, a, false, b, (short)0, c,
                                                   false, false);
}

// fast silu: v * sigmoid(v), single v_exp_f32 + v_rcp_f32 (no IEEE div sequence)
__device__ __forceinline__ float fast_silu(float v) {
    return v * __builtin_amdgcn_rcpf(1.f + __expf(-v));
}

// A fragment (16x32 bf16, ISA layout) from LDS.
// lane 0-15 : row m=lane, K = k0 + {0..7, 16..23}
// lane16-31 : row m=lane-16, K = k0 + {8..15, 24..31}
__device__ __forceinline__ v16bf load_a_frag(const __bf16* base, int lane, int rs) {
    int m  = lane & 15;
    int kb = (lane >> 4) * 8;
    const __bf16* p = base + m * rs + kb;
    v16bf a;
#pragma unroll
    for (int i = 0; i < 8; ++i) a[i] = p[i];          // -> ds_load_b128
#pragma unroll
    for (int i = 0; i < 8; ++i) a[8 + i] = p[16 + i]; // -> ds_load_b128
    return a;
}

// B fragment (32x16 bf16, ISA layout) from global fp32 row-major [K x N], leading dim ld.
// lane 0-15 : col n=lane,    K = k0 + {0..7, 16..23}
// lane16-31 : col n=lane-16, K = k0 + {8..15, 24..31}
// For fixed i, lanes 0-15 read 16 consecutive fp32 -> 64B coalesced segments.
__device__ __forceinline__ v16bf load_b_frag(const float* p0, int lane, int ld) {
    int n  = lane & 15;
    int kb = (lane >> 4) * 8;
    const float* p = p0 + (size_t)kb * ld + n;
    v16bf b;
#pragma unroll
    for (int i = 0; i < 16; ++i) b[i] = (__bf16)p[(size_t)(i < 8 ? i : i + 8) * ld];
    return b;
}

// ---------------------------------------------------------------- init
__global__ void moe_init_kernel(int* counts, int* cursor) {
    int i = threadIdx.x;
    if (i < EE) { counts[i] = 0; cursor[i] = 0; }
}

// ---------------------------------------------------------------- router
// one wave32 per token
__global__ __launch_bounds__(NTHREADS) void moe_router_kernel(
    const float* __restrict__ x, const float* __restrict__ rw,
    const float* __restrict__ rb, int* __restrict__ top1,
    float* __restrict__ prob, float* __restrict__ probs8,
    int* __restrict__ counts) {
    int wave = threadIdx.x >> 5;
    int lane = threadIdx.x & 31;
    int t = blockIdx.x * NWAVES + wave;
    if (t >= TB) return;

    const float* xr = x + (size_t)t * CC;
    float acc[EE];
#pragma unroll
    for (int e = 0; e < EE; ++e) acc[e] = 0.f;
    for (int c = lane; c < CC; c += 32) {
        float xv = xr[c];
        const float* wr = rw + (size_t)c * EE;
#pragma unroll
        for (int e = 0; e < EE; ++e) acc[e] += xv * wr[e];
    }
#pragma unroll
    for (int e = 0; e < EE; ++e) {
#pragma unroll
        for (int s = 16; s > 0; s >>= 1) acc[e] += __shfl_xor(acc[e], s, 32);
    }
    if (lane == 0) {
        float lg[EE];
        float mx = -1e30f;
#pragma unroll
        for (int e = 0; e < EE; ++e) { lg[e] = acc[e] + rb[e]; mx = fmaxf(mx, lg[e]); }
        float den = 0.f;
#pragma unroll
        for (int e = 0; e < EE; ++e) { lg[e] = __expf(lg[e] - mx); den += lg[e]; }
        float inv = 1.f / den;   // one-off, keep precise
        int am = 0; float pm = -1.f;
#pragma unroll
        for (int e = 0; e < EE; ++e) {
            float p = lg[e] * inv;
            probs8[(size_t)t * EE + e] = p;
            if (p > pm) { pm = p; am = e; }   // first-max tie-break like argmax
        }
        top1[t] = am;
        prob[t] = pm;
        atomicAdd(&counts[am], 1);
    }
}

// ---------------------------------------------------------------- scan + aux
// deterministic reduction of importance; load = counts/T exactly
__global__ __launch_bounds__(NTHREADS) void moe_scan_aux_kernel(
    const int* __restrict__ counts, int* __restrict__ offsets,
    const float* __restrict__ probs8, float* __restrict__ aux_out) {
    __shared__ float red[NTHREADS];
    __shared__ float imp[EE];
    for (int e = 0; e < EE; ++e) {
        float s = 0.f;
        for (int t = threadIdx.x; t < TB; t += NTHREADS) s += probs8[(size_t)t * EE + e];
        red[threadIdx.x] = s;
        __syncthreads();
        for (int st = NTHREADS / 2; st > 0; st >>= 1) {
            if (threadIdx.x < st) red[threadIdx.x] += red[threadIdx.x + st];
            __syncthreads();
        }
        if (threadIdx.x == 0) imp[e] = red[0] / (float)TB;
        __syncthreads();
    }
    if (threadIdx.x == 0) {
        int acc = 0; float aux = 0.f;
        for (int e = 0; e < EE; ++e) {
            offsets[e] = acc; acc += counts[e];
            aux += imp[e] * ((float)counts[e] / (float)TB);
        }
        *aux_out = (float)EE * aux;
    }
}

// ---------------------------------------------------------------- scatter
__global__ __launch_bounds__(NTHREADS) void moe_scatter_kernel(
    const int* __restrict__ top1, const int* __restrict__ offsets,
    int* __restrict__ cursor, int* __restrict__ order) {
    int t = blockIdx.x * NTHREADS + threadIdx.x;
    if (t >= TB) return;
    int e = top1[t];
    int pos = offsets[e] + atomicAdd(&cursor[e], 1);
    order[pos] = t;
}

// ---------------------------------------------------------------- expert FFN
// one block = one 16-token tile of one expert; 8 waves, dual-N-tile WMMA bf16
__global__ __launch_bounds__(NTHREADS) void moe_ffn_kernel(
    const float* __restrict__ x,  const float* __restrict__ w1,
    const float* __restrict__ b1, const float* __restrict__ w2,
    const float* __restrict__ b2, const int* __restrict__ counts,
    const int* __restrict__ offsets, const int* __restrict__ order,
    const float* __restrict__ prob, float* __restrict__ out) {
    extern __shared__ char smem[];
    __bf16* xs = (__bf16*)smem;                // [16][XS_STRIDE]
    __bf16* hs = (__bf16*)(smem + XS_BYTES);   // [16][HS_STRIDE]
    __shared__ int   sMeta[3];                 // expert, tile, count
    __shared__ int   sRowTok[16];
    __shared__ float sRowP[16];

    if (threadIdx.x == 0) {
        int b = blockIdx.x, acc = 0, e = -1, tile = 0;
        for (int i = 0; i < EE; ++i) {
            int tl = (counts[i] + 15) >> 4;
            if (e < 0 && b < acc + tl) { e = i; tile = b - acc; }
            acc += tl;
        }
        sMeta[0] = e; sMeta[1] = tile; sMeta[2] = (e >= 0) ? counts[e] : 0;
    }
    __syncthreads();
    int e = sMeta[0];
    if (e < 0) return;                       // uniform for the whole block
    int tile = sMeta[1];
    int cnt  = sMeta[2];
    int off  = offsets[e];

    if (threadIdx.x < 16) {
        int idx = tile * 16 + threadIdx.x;
        int t = (idx < cnt) ? order[off + idx] : 0;
        sRowTok[threadIdx.x] = t;
        sRowP[threadIdx.x]   = (idx < cnt) ? prob[t] : 0.f;
    }
    __syncthreads();

    // stage x tile -> bf16 LDS (coalesced)
    for (int i = threadIdx.x; i < 16 * CC; i += NTHREADS) {
        int m = i >> 10, c = i & (CC - 1);
        xs[m * XS_STRIDE + c] = (__bf16)x[(size_t)sRowTok[m] * CC + c];
    }
    __syncthreads();

    int wave = threadIdx.x >> 5;
    int lane = threadIdx.x & 31;
    const float* w1e = w1 + (size_t)e * CC * FF;
    const float* b1e = b1 + (size_t)e * FF;
    const float* w2e = w2 + (size_t)e * FF * CC;
    const float* b2e = b2 + (size_t)e * CC;

    // ---- phase 1: h[16,FF] = silu(x @ w1 + b1) ----
    // each wave owns pairs of adjacent 16-col tiles (32 cols); A-frag shared by 2 WMMAs
    for (int p = wave; p < FF / 32; p += NWAVES) {   // 16 iterations per wave
        v8f acc0 = {}, acc1 = {};
        const float* wb0 = w1e + p * 32;
        const float* wb1 = wb0 + 16;
        for (int k0 = 0; k0 < CC; k0 += 32) {
            __builtin_prefetch(wb0 + (size_t)(k0 + 32) * FF, 0, 1);
            v16bf a  = load_a_frag(xs + k0, lane, XS_STRIDE);
            v16bf b0 = load_b_frag(wb0 + (size_t)k0 * FF, lane, FF);
            v16bf b1f = load_b_frag(wb1 + (size_t)k0 * FF, lane, FF);
            acc0 = wmma_bf16(a, b0, acc0);
            acc1 = wmma_bf16(a, b1f, acc1);
        }
        int n0 = p * 32 + (lane & 15);
        int n1 = n0 + 16;
        float bias0 = b1e[n0];
        float bias1 = b1e[n1];
        int mbase = (lane >> 4) * 8;
#pragma unroll
        for (int j = 0; j < 8; ++j) {
            int m = mbase + j;
            hs[m * HS_STRIDE + n0] = (__bf16)fast_silu(acc0[j] + bias0);
            hs[m * HS_STRIDE + n1] = (__bf16)fast_silu(acc1[j] + bias1);
        }
    }
    __syncthreads();

    // ---- phase 2: y[16,CC] = h @ w2 + b2, scaled by top1 prob ----
    for (int p = wave; p < CC / 32; p += NWAVES) {   // 4 iterations per wave
        v8f acc0 = {}, acc1 = {};
        const float* wb0 = w2e + p * 32;
        const float* wb1 = wb0 + 16;
        for (int k0 = 0; k0 < FF; k0 += 32) {
            __builtin_prefetch(wb0 + (size_t)(k0 + 32) * CC, 0, 1);
            v16bf a  = load_a_frag(hs + k0, lane, HS_STRIDE);
            v16bf b0 = load_b_frag(wb0 + (size_t)k0 * CC, lane, CC);
            v16bf b1f = load_b_frag(wb1 + (size_t)k0 * CC, lane, CC);
            acc0 = wmma_bf16(a, b0, acc0);
            acc1 = wmma_bf16(a, b1f, acc1);
        }
        int n0 = p * 32 + (lane & 15);
        int n1 = n0 + 16;
        float bias0 = b2e[n0];
        float bias1 = b2e[n1];
        int mbase = (lane >> 4) * 8;
#pragma unroll
        for (int j = 0; j < 8; ++j) {
            int m = mbase + j;
            if (tile * 16 + m < cnt) {
                int t = sRowTok[m];
                float pscale = sRowP[m];
                out[(size_t)t * CC + n0] = (acc0[j] + bias0) * pscale;
                out[(size_t)t * CC + n1] = (acc1[j] + bias1) * pscale;
            }
        }
    }
}

// ---------------------------------------------------------------- launch
extern "C" void kernel_launch(void* const* d_in, const int* in_sizes, int n_in,
                              void* d_out, int out_size, void* d_ws, size_t ws_size,
                              hipStream_t stream) {
    const float* x  = (const float*)d_in[0];
    const float* rw = (const float*)d_in[1];
    const float* rb = (const float*)d_in[2];
    const float* w1 = (const float*)d_in[3];
    const float* b1 = (const float*)d_in[4];
    const float* w2 = (const float*)d_in[5];
    const float* b2 = (const float*)d_in[6];
    float* out = (float*)d_out;
    float* aux = out + (size_t)TB * CC;

    // workspace layout
    char* ws = (char*)d_ws;
    int*   counts  = (int*)ws;                  // [8]
    int*   cursor  = counts + 8;                // [8]
    int*   offsets = cursor + 8;                // [8]
    int*   top1    = offsets + 8;               // [TB]
    int*   order   = top1 + TB;                 // [TB]
    float* prob    = (float*)(order + TB);      // [TB]
    float* probs8  = prob + TB;                 // [TB*EE]

    moe_init_kernel<<<1, 32, 0, stream>>>(counts, cursor);
    moe_router_kernel<<<TB / NWAVES, NTHREADS, 0, stream>>>(
        x, rw, rb, top1, prob, probs8, counts);
    moe_scan_aux_kernel<<<1, NTHREADS, 0, stream>>>(counts, offsets, probs8, aux);
    moe_scatter_kernel<<<TB / NTHREADS, NTHREADS, 0, stream>>>(
        top1, offsets, cursor, order);
    moe_ffn_kernel<<<TB / 16 + EE, NTHREADS, SMEM_BYTES, stream>>>(
        x, w1, b1, w2, b2, counts, offsets, order, prob, out);
}
